// Attention_18726057410905
// MI455X (gfx1250) — compile-verified, matrix operand order checked by reading, not run on hardware
//
#include <hip/hip_runtime.h>

// ---------------------------------------------------------------------------
// MI455X (gfx1250) multi-head attention forward.
// B=8, S=1024, D=768, N=12, H=64.  bf16 WMMA (16x16x32) with f32 accumulate.
// Global->LDS staging via Tensor Data Mover (TENSORcnt), double-buffered GEMM.
// ---------------------------------------------------------------------------

typedef __attribute__((ext_vector_type(16))) __bf16 v16bf;
typedef __attribute__((ext_vector_type(8)))  float  v8f;
typedef __attribute__((ext_vector_type(4)))  unsigned int v4u;
typedef __attribute__((ext_vector_type(8)))  int v8i;
typedef __attribute__((ext_vector_type(4)))  int v4i;

#if defined(__has_builtin)
#if __has_builtin(__builtin_amdgcn_tensor_load_to_lds) && \
    __has_builtin(__builtin_amdgcn_s_wait_tensorcnt)
#define USE_TDM 1
#endif
#endif
#ifndef USE_TDM
#define USE_TDM 0
#endif

struct __attribute__((aligned(16))) U128 { unsigned int x, y, z, w; };
union V16U { U128 u[2]; v16bf v; unsigned short s[16]; };
union U128S { U128 u; unsigned short s[8]; };

__device__ __forceinline__ unsigned short f2bf(float f) {
  unsigned int u = __float_as_uint(f);
  u += 0x7FFFu + ((u >> 16) & 1u);           // round-to-nearest-even
  return (unsigned short)(u >> 16);
}

// Load one 16x(32-K-chunk) fragment: two contiguous 8-elem (16B) runs per lane.
__device__ __forceinline__ v16bf ld_frag(const unsigned short* lo,
                                         const unsigned short* hi) {
  V16U t;
  t.u[0] = *(const U128*)lo;
  t.u[1] = *(const U128*)hi;
  return t.v;
}

#if USE_TDM
// TDM: DMA a 2-D bf16 tile [tile_h rows x tile_w elems] (row stride
// row_stride_elems) from global into LDS at lds_addr, inserting
// pad DWORDs per interval so the LDS row stride matches our padded layout.
// D# fields per CDNA5 ISA.
// This toolchain exposes the 6-operand builtin form:
//   (uint32x4 g0, int32x8 g1, int32x4 g2, int32x4 g3, int32x8, i32 cpol)
__device__ __forceinline__ void tdm_load_2d(unsigned lds_addr,
                                            const void* gptr,
                                            unsigned tile_w, unsigned tile_h,
                                            unsigned row_stride_elems,
                                            unsigned pad_interval_code,
                                            unsigned pad_amount_code) {
  unsigned long long ga = (unsigned long long)(uintptr_t)gptr;
  v4u g0;
  g0[0] = 1u;                                   // count=1 valid, user desc
  g0[1] = lds_addr;                             // LDS byte address
  g0[2] = (unsigned)ga;                         // global addr [31:0]
  g0[3] = (unsigned)(ga >> 32) | (2u << 30);    // addr [56:32] | type=2
  v8i g1;
  g1[0] = (int)((1u << 16)                      // data_size=1 -> 2 bytes
                | (1u << 20)                    // pad_enable
                | (pad_interval_code << 22)
                | (pad_amount_code << 25));
  g1[1] = (int)((tile_w & 0xFFFFu) << 16);      // tensor_dim0[15:0]
  g1[2] = (int)((tile_w >> 16) | ((tile_h & 0xFFFFu) << 16)); // d0 hi | d1 lo
  g1[3] = (int)((tile_h >> 16) | (tile_w << 16));             // d1 hi | tile_dim0
  g1[4] = (int)(tile_h & 0xFFFFu);              // tile_dim1 | tile_dim2=0
  g1[5] = (int)row_stride_elems;                // tensor_dim0_stride[31:0]
  g1[6] = 0;                                    // stride hi | dim1_stride lo
  g1[7] = 0;
  v4i g2 = {0, 0, 0, 0};                        // 2-D: dims 2..4 unused
  v4i g3 = {0, 0, 0, 0};
  v8i g4 = {0, 0, 0, 0, 0, 0, 0, 0};            // unused trailing group
  __builtin_amdgcn_tensor_load_to_lds(g0, g1, g2, g3, g4, 0);
}
#endif

// ------------------------------ pack kernels -------------------------------

__global__ __launch_bounds__(256)
void cvt_f32_bf16_kernel(const float* __restrict__ in,
                         unsigned short* __restrict__ out, int n) {
  int i = blockIdx.x * 256 + threadIdx.x;
  if (i < n) out[i] = f2bf(in[i]);
}

// W[N][D][H] (f32) -> out[(n*H+h)][d] (bf16), i.e. B^T layout for x @ W
__global__ __launch_bounds__(256)
void pack_w_dh_kernel(const float* __restrict__ W,
                      unsigned short* __restrict__ out) {
  int o = blockIdx.x * 256 + threadIdx.x;
  if (o >= 768 * 768) return;
  int nh = o / 768, d = o % 768;
  int n = nh >> 6, h = nh & 63;
  out[o] = f2bf(W[((size_t)n * 768 + d) * 64 + h]);
}

// W_O[N][H][D] == [nh][d] (f32) -> out[d][nh] (bf16), B^T layout for z @ W_O
__global__ __launch_bounds__(256)
void pack_wo_kernel(const float* __restrict__ W,
                    unsigned short* __restrict__ out) {
  int o = blockIdx.x * 256 + threadIdx.x;
  if (o >= 768 * 768) return;
  int d = o / 768, nh = o % 768;
  out[o] = f2bf(W[(size_t)nh * 768 + d]);
}

// ------------------------------ tiled GEMM ---------------------------------
// C[M][N] = A[M][K] * B[K][N] + bias[N], with B given as BT[N][K] (bf16).
// Block: 256 threads = 8 waves; tile 128(M) x 64(N), BK=32.
// Wave (wm,wn) owns a 32x32 sub-tile = 2x2 WMMA fragments.
// Staging: TDM double-buffer (wave 0 DMAs tile k+1 while all waves WMMA on k).

template <bool OUT_BF16>
__global__ __launch_bounds__(256)
void gemm_bf16_kernel(const unsigned short* __restrict__ A,
                      const unsigned short* __restrict__ BT,
                      const float* __restrict__ bias,
                      void* __restrict__ Cout,
                      int M, int N, int K) {
  __shared__ unsigned short sA[2][128 * 40];  // 64B row + 16B pad (16B aligned)
  __shared__ unsigned short sB[2][64 * 40];

  const int tid  = threadIdx.x;
  const int lane = tid & 31;
  const int wid  = tid >> 5;
  const int g    = lane >> 4;      // half-wave group
  const int ln   = lane & 15;      // M (A) / N (B,C) index within fragment
  const int wm   = wid & 3;        // 0..3  -> M offset wm*32
  const int wn   = wid >> 2;       // 0..1  -> N offset wn*32
  const int m0   = blockIdx.y * 128;
  const int n0   = blockIdx.x * 64;

  v8f acc[2][2];
#pragma unroll
  for (int i = 0; i < 2; ++i)
#pragma unroll
    for (int j = 0; j < 2; ++j)
#pragma unroll
      for (int r = 0; r < 8; ++r) acc[i][j][r] = 0.f;

#if USE_TDM
  const bool tdmw = (wid == 0);
  const unsigned ldsA[2] = {(unsigned)(uintptr_t)&sA[0][0],
                            (unsigned)(uintptr_t)&sA[1][0]};
  const unsigned ldsB[2] = {(unsigned)(uintptr_t)&sB[0][0],
                            (unsigned)(uintptr_t)&sB[1][0]};
  if (tdmw) {
    // rows: 32 elems = 16 DW -> interval code 3; pad 4 DW -> amount code 3
    tdm_load_2d(ldsA[0], A + (size_t)m0 * K, 32, 128, (unsigned)K, 3, 3);
    tdm_load_2d(ldsB[0], BT + (size_t)n0 * K, 32, 64, (unsigned)K, 3, 3);
  }
#endif

  int buf = 0;
  for (int kk = 0; kk < K; kk += 32) {
#if USE_TDM
    if (tdmw) {
      if (kk + 32 < K) {
        tdm_load_2d(ldsA[buf ^ 1], A + (size_t)m0 * K + (kk + 32), 32, 128,
                    (unsigned)K, 3, 3);
        tdm_load_2d(ldsB[buf ^ 1], BT + (size_t)n0 * K + (kk + 32), 32, 64,
                    (unsigned)K, 3, 3);
        __builtin_amdgcn_s_wait_tensorcnt((short)2);  // current tile done
      } else {
        __builtin_amdgcn_s_wait_tensorcnt((short)0);
      }
    }
    __syncthreads();
#else
    // manual staging fallback
#pragma unroll
    for (int i = 0; i < 2; ++i) {
      int idx = tid + i * 256;
      int r = idx >> 2, cc = idx & 3;
      *(U128*)&sA[0][r * 40 + cc * 8] =
          *(const U128*)(A + (size_t)(m0 + r) * K + kk + cc * 8);
    }
    {
      int r = tid >> 2, cc = tid & 3;
      *(U128*)&sB[0][r * 40 + cc * 8] =
          *(const U128*)(BT + (size_t)(n0 + r) * K + kk + cc * 8);
    }
    __syncthreads();
#endif

    v16bf af[2], bfr[2];
#pragma unroll
    for (int i = 0; i < 2; ++i) {
      int mr = wm * 32 + i * 16 + ln;
      af[i] = ld_frag(&sA[buf][mr * 40 + 8 * g],
                      &sA[buf][mr * 40 + 16 + 8 * g]);
    }
#pragma unroll
    for (int j = 0; j < 2; ++j) {
      int nr = wn * 32 + j * 16 + ln;
      bfr[j] = ld_frag(&sB[buf][nr * 40 + 8 * g],
                       &sB[buf][nr * 40 + 16 + 8 * g]);
    }
#pragma unroll
    for (int i = 0; i < 2; ++i)
#pragma unroll
      for (int j = 0; j < 2; ++j)
        acc[i][j] = __builtin_amdgcn_wmma_f32_16x16x32_bf16(
            false, af[i], false, bfr[j], (short)0, acc[i][j], false, false);
    __syncthreads();
#if USE_TDM
    buf ^= 1;
#endif
  }

  // epilogue: C layout (M = r + 8g, N = ln)
#pragma unroll
  for (int i = 0; i < 2; ++i)
#pragma unroll
    for (int j = 0; j < 2; ++j)
#pragma unroll
      for (int r = 0; r < 8; ++r) {
        int m = m0 + wm * 32 + i * 16 + r + 8 * g;
        int c = n0 + wn * 32 + j * 16 + ln;
        float v = acc[i][j][r] + bias[c];
        if (OUT_BF16)
          ((unsigned short*)Cout)[(size_t)m * N + c] = f2bf(v);
        else
          ((float*)Cout)[(size_t)m * N + c] = v;
      }
}

// ---------------------------- flash attention ------------------------------
// Grid: (qb=16, head=12, batch=8).  Block: 128 threads = 4 waves.
// Wave w owns queries q0 + w*16 .. +15; loops over 64-key blocks (causal).
// K tile staged by TDM; V tile transposed manually (TDM cannot transpose).

__global__ __launch_bounds__(128)
void flash_attn_kernel(const unsigned short* __restrict__ Qb,
                       const unsigned short* __restrict__ Kb,
                       const unsigned short* __restrict__ Vb,
                       unsigned short* __restrict__ Zb) {
  __shared__ unsigned short sK[64 * 72];      // [key][h]  (128B row + 16B pad)
  __shared__ unsigned short sVt[64 * 72];     // [h][key]  (transposed V)
  __shared__ unsigned short sP[4 * 16 * 72];  // per-wave P staging

  const int tid  = threadIdx.x;
  const int lane = tid & 31;
  const int w    = tid >> 5;
  const int g    = lane >> 4;
  const int ln   = lane & 15;
  const int qb   = blockIdx.x;     // query block 0..15
  const int n    = blockIdx.y;     // head
  const int b    = blockIdx.z;     // batch
  const int q0   = qb * 64;
  const size_t rowN = 768;         // N*H
  const size_t base = (size_t)b * 1024 * rowN + (size_t)n * 64;

  // Q fragments, kept in registers for the whole loop (H=64 -> 2 K-chunks)
  v16bf qf[2];
  {
    const unsigned short* qrow = Qb + base + (size_t)(q0 + w * 16 + ln) * rowN;
#pragma unroll
    for (int c = 0; c < 2; ++c)
      qf[c] = ld_frag(qrow + c * 32 + 8 * g, qrow + c * 32 + 16 + 8 * g);
  }

  float row_max[8], row_sum[8];
  v8f acco[4];
#pragma unroll
  for (int r = 0; r < 8; ++r) { row_max[r] = -1e30f; row_sum[r] = 0.f; }
#pragma unroll
  for (int f = 0; f < 4; ++f)
#pragma unroll
    for (int r = 0; r < 8; ++r) acco[f][r] = 0.f;

  unsigned short* sPw = &sP[w * 16 * 72];
#if USE_TDM
  const unsigned ldsK = (unsigned)(uintptr_t)&sK[0];
#endif

  for (int kb = 0; kb <= qb; ++kb) {
#if USE_TDM
    if (w == 0)  // rows: 64 elems = 32 DW -> interval code 4; pad 4 DW -> 3
      tdm_load_2d(ldsK, Kb + base + (size_t)(kb * 64) * rowN, 64, 64, 768, 4, 3);
#endif
    // V tile: load row-major, store transposed into LDS (all threads)
#pragma unroll
    for (int i = 0; i < 4; ++i) {
      int idx = tid + i * 128;            // 0..511
      int r = idx >> 3, cc = idx & 7;     // key row, 8-col chunk
      const size_t grow = base + (size_t)(kb * 64 + r) * rowN + cc * 8;
#if !USE_TDM
      *(U128*)&sK[r * 72 + cc * 8] = *(const U128*)(Kb + grow);
#endif
      U128S vv; vv.u = *(const U128*)(Vb + grow);
#pragma unroll
      for (int e = 0; e < 8; ++e) sVt[(cc * 8 + e) * 72 + r] = vv.s[e];
    }
#if USE_TDM
    if (w == 0) __builtin_amdgcn_s_wait_tensorcnt((short)0);
#endif
    __syncthreads();

    // scores: S = Q * K^T   (4 key fragments x 2 H-chunks)
    v8f sc[4];
#pragma unroll
    for (int j = 0; j < 4; ++j)
#pragma unroll
      for (int r = 0; r < 8; ++r) sc[j][r] = 0.f;
#pragma unroll
    for (int c = 0; c < 2; ++c)
#pragma unroll
      for (int j = 0; j < 4; ++j) {
        int kr = j * 16 + ln;
        v16bf bk = ld_frag(&sK[kr * 72 + c * 32 + 8 * g],
                           &sK[kr * 72 + c * 32 + 16 + 8 * g]);
        sc[j] = __builtin_amdgcn_wmma_f32_16x16x32_bf16(
            false, qf[c], false, bk, (short)0, sc[j], false, false);
      }

    // scale 1/sqrt(64), causal mask, online softmax
#pragma unroll
    for (int r = 0; r < 8; ++r) {
      int qi = q0 + w * 16 + r + 8 * g;
      float mx = -1e30f;
#pragma unroll
      for (int j = 0; j < 4; ++j) {
        int ki = kb * 64 + j * 16 + ln;
        float s = sc[j][r] * 0.125f;
        if (ki > qi) s = -1e30f;
        sc[j][r] = s;
        mx = fmaxf(mx, s);
      }
#pragma unroll
      for (int off = 8; off; off >>= 1)
        mx = fmaxf(mx, __shfl_xor(mx, off, 32));
      float mnew = fmaxf(row_max[r], mx);
      float scale = __expf(row_max[r] - mnew);
      float rs = 0.f;
#pragma unroll
      for (int j = 0; j < 4; ++j) {
        float p = __expf(sc[j][r] - mnew);
        sc[j][r] = p;
        rs += p;
      }
#pragma unroll
      for (int off = 8; off; off >>= 1) rs += __shfl_xor(rs, off, 32);
      row_sum[r] = row_sum[r] * scale + rs;
      row_max[r] = mnew;
#pragma unroll
      for (int f = 0; f < 4; ++f) acco[f][r] *= scale;
    }

    // C-layout -> A-layout via per-wave LDS (bf16)
#pragma unroll
    for (int j = 0; j < 4; ++j)
#pragma unroll
      for (int r = 0; r < 8; ++r)
        sPw[(r + 8 * g) * 72 + j * 16 + ln] = f2bf(sc[j][r]);

    // O += P * V   (2 key-chunks x 4 H fragments)
#pragma unroll
    for (int c = 0; c < 2; ++c) {
      v16bf pa = ld_frag(&sPw[ln * 72 + c * 32 + 8 * g],
                         &sPw[ln * 72 + c * 32 + 16 + 8 * g]);
#pragma unroll
      for (int f = 0; f < 4; ++f) {
        int hr = f * 16 + ln;
        v16bf vb = ld_frag(&sVt[hr * 72 + c * 32 + 8 * g],
                           &sVt[hr * 72 + c * 32 + 16 + 8 * g]);
        acco[f] = __builtin_amdgcn_wmma_f32_16x16x32_bf16(
            false, pa, false, vb, (short)0, acco[f], false, false);
      }
    }
    __syncthreads();
  }

  // normalize + store Z (bf16)
  {
    size_t p = (size_t)b * 1024 + q0 + w * 16;
#pragma unroll
    for (int r = 0; r < 8; ++r) {
      float inv = 1.0f / row_sum[r];
      size_t row = (p + r + 8 * g) * rowN + (size_t)n * 64;
#pragma unroll
      for (int f = 0; f < 4; ++f)
        Zb[row + f * 16 + ln] = f2bf(acco[f][r] * inv);
    }
  }
}

// ------------------------------- launcher ----------------------------------

extern "C" void kernel_launch(void* const* d_in, const int* in_sizes, int n_in,
                              void* d_out, int out_size, void* d_ws,
                              size_t ws_size, hipStream_t stream) {
  (void)in_sizes; (void)n_in; (void)out_size; (void)ws_size;

  const float* x   = (const float*)d_in[0];
  const float* W_Q = (const float*)d_in[1];
  const float* W_K = (const float*)d_in[2];
  const float* W_V = (const float*)d_in[3];
  const float* W_O = (const float*)d_in[4];
  const float* b_Q = (const float*)d_in[5];
  const float* b_K = (const float*)d_in[6];
  const float* b_V = (const float*)d_in[7];
  const float* b_O = (const float*)d_in[8];

  const size_t SZ_X = (size_t)8192 * 768;   // activations (elements)
  const size_t SZ_W = (size_t)768 * 768;    // packed weights (elements)

  unsigned short* ws  = (unsigned short*)d_ws;
  unsigned short* xb  = ws;
  unsigned short* WqT = xb + SZ_X;
  unsigned short* WkT = WqT + SZ_W;
  unsigned short* WvT = WkT + SZ_W;
  unsigned short* WoT = WvT + SZ_W;
  unsigned short* Qb  = WoT + SZ_W;
  unsigned short* Kb  = Qb + SZ_X;
  unsigned short* Vb  = Kb + SZ_X;
  unsigned short* Zb  = Vb + SZ_X;

  cvt_f32_bf16_kernel<<<(int)((SZ_X + 255) / 256), 256, 0, stream>>>(
      x, xb, (int)SZ_X);
  const int wblocks = (int)((SZ_W + 255) / 256);
  pack_w_dh_kernel<<<wblocks, 256, 0, stream>>>(W_Q, WqT);
  pack_w_dh_kernel<<<wblocks, 256, 0, stream>>>(W_K, WkT);
  pack_w_dh_kernel<<<wblocks, 256, 0, stream>>>(W_V, WvT);
  pack_wo_kernel<<<wblocks, 256, 0, stream>>>(W_O, WoT);

  dim3 ggrid(12, 64);  // N/64, M/128
  gemm_bf16_kernel<true><<<ggrid, 256, 0, stream>>>(xb, WqT, b_Q, Qb, 8192, 768, 768);
  gemm_bf16_kernel<true><<<ggrid, 256, 0, stream>>>(xb, WkT, b_K, Kb, 8192, 768, 768);
  gemm_bf16_kernel<true><<<ggrid, 256, 0, stream>>>(xb, WvT, b_V, Vb, 8192, 768, 768);

  flash_attn_kernel<<<dim3(16, 12, 8), 128, 0, stream>>>(Qb, Kb, Vb, Zb);

  gemm_bf16_kernel<false><<<ggrid, 256, 0, stream>>>(Zb, WoT, b_O, d_out,
                                                     8192, 768, 768);
}
